// Drone1BitCNN_17489106829367
// MI455X (gfx1250) — 8-bit matrix operand order repaired and checked by reading, compiled, not run on hardware
//
#include <hip/hip_runtime.h>

typedef __attribute__((ext_vector_type(8))) int v8i;

#define EPS 1e-5f

#define OFF_SGN   0
#define OFF_P2    4368
#define OFF_P3    0
#define OFF_ACT1  32768
#define OFF_ACT2  42016
#define OFF_H     47200
#define OFF_RED   55392
#define OFF_CONST 57440
#define SMEM_BYTES 58048

__global__ __launch_bounds__(256) void drone1bit_fused(
    const float* __restrict__ x,
    const float* __restrict__ w1, const float* __restrict__ b1,
    const float* __restrict__ g1, const float* __restrict__ be1,
    const float* __restrict__ m1, const float* __restrict__ v1,
    const float* __restrict__ w2, const float* __restrict__ b2,
    const float* __restrict__ g2, const float* __restrict__ be2,
    const float* __restrict__ m2, const float* __restrict__ v2,
    const float* __restrict__ w3, const float* __restrict__ b3,
    const float* __restrict__ g3, const float* __restrict__ be3,
    const float* __restrict__ m3, const float* __restrict__ v3,
    const float* __restrict__ fcw, const float* __restrict__ fcb,
    float* __restrict__ out)
{
    __shared__ __align__(16) unsigned char smem[SMEM_BYTES];
    const int tid  = threadIdx.x;
    const int img  = blockIdx.x;
    const int lane = tid & 31;
    const int wave = tid >> 5;
    const int hi   = lane >> 4;
    const int ncol = lane & 15;
    const int mlo  = lane & 15;

    float* cf     = (float*)(smem + OFF_CONST);
    float* scale1 = cf;        float* shift1 = cf + 8;
    float* scale2 = cf + 16;   float* shift2 = cf + 32;
    float* scale3 = cf + 48;   float* shift3 = cf + 80;
    signed char* w1s = (signed char*)(smem + OFF_CONST + 448);

    for (int i = tid; i < 4356; i += 256) smem[OFF_SGN  + i] = 0;
    for (int i = tid; i < 9248; i += 256) smem[OFF_ACT1 + i] = 0;
    for (int i = tid; i < 5184; i += 256) smem[OFF_ACT2 + i] = 0;
    if (tid < 8)  { float inv = g1[tid]*rsqrtf(v1[tid]+EPS); scale1[tid]=inv; shift1[tid]=b1[tid]*inv + be1[tid] - m1[tid]*inv; }
    if (tid < 16) { float inv = g2[tid]*rsqrtf(v2[tid]+EPS); scale2[tid]=inv; shift2[tid]=b2[tid]*inv + be2[tid] - m2[tid]*inv; }
    if (tid < 32) { float inv = g3[tid]*rsqrtf(v3[tid]+EPS); scale3[tid]=inv; shift3[tid]=b3[tid]*inv + be3[tid] - m3[tid]*inv; }
    if (tid < 72) { float w = w1[tid]; w1s[tid] = (signed char)((w>0.f)-(w<0.f)); }
    __syncthreads();

    {
        const float4* xin = (const float4*)(x + (size_t)img * 4096);
        signed char* sg = (signed char*)(smem + OFF_SGN);
        for (int i4 = tid; i4 < 1024; i4 += 256) {
            float4 v = xin[i4];
            int i = i4 << 2;
            signed char* dst = sg + ((i >> 6) + 1) * 66 + (i & 63) + 1;
            dst[0] = (signed char)((v.x>0.f)-(v.x<0.f));
            dst[1] = (signed char)((v.y>0.f)-(v.y<0.f));
            dst[2] = (signed char)((v.z>0.f)-(v.z<0.f));
            dst[3] = (signed char)((v.w>0.f)-(v.w<0.f));
        }
    }
    __syncthreads();

    {
        const signed char* sg = (const signed char*)(smem + OFF_SGN);
        unsigned char* a1 = smem + OFF_ACT1;
        for (int p = tid; p < 1024; p += 256) {
            int py = p >> 5, px = p & 31;
            unsigned mask = 0;
            for (int a = 0; a < 2; ++a)
            for (int b = 0; b < 2; ++b) {
                int cy = 2*py + a, cx = 2*px + b;
                int s[9];
                #pragma unroll
                for (int t = 0; t < 9; ++t)
                    s[t] = sg[(cy + t/3)*66 + (cx + t%3)];
                #pragma unroll
                for (int c = 0; c < 8; ++c) {
                    int sum = 0;
                    #pragma unroll
                    for (int t = 0; t < 9; ++t) sum += s[t] * (int)w1s[c*9 + t];
                    float bn = (float)sum * scale1[c] + shift1[c];
                    if (bn > 0.f) mask |= 1u << c;
                }
            }
            #pragma unroll
            for (int c = 0; c < 8; ++c)
                a1[((py+1)*34 + (px+1))*8 + c] = (mask >> c) & 1;
        }
    }
    __syncthreads();

    {
        v8i bf[2];
        for (int q = 0; q < 2; ++q)
            for (int v = 0; v < 8; ++v) {
                unsigned d = 0;
                for (int j = 0; j < 4; ++j) {
                    int k = 64*q + 16*(2*(v>>2) + hi) + ((v&3)<<2) + j;
                    int val = 0;
                    if (k < 72) {
                        int tap = k >> 3, cin = k & 7;
                        float w = w2[ncol*72 + cin*9 + tap];
                        val = (w>0.f)-(w<0.f);
                    }
                    d |= ((unsigned)(val & 0xFF)) << (8*j);
                }
                bf[q][v] = (int)d;
            }

        const unsigned* a1w = (const unsigned*)(smem + OFF_ACT1);
        unsigned char* p2 = smem + OFF_P2;
        const float sc = scale2[ncol], sh = shift2[ncol];
        for (int t0 = wave*8; t0 < wave*8 + 8; t0 += 4) {
            v8i acc[4], A[4];
            #pragma unroll
            for (int u = 0; u < 4; ++u) {
                int mA = (t0+u)*16 + mlo;
                int cy = mA >> 5, cx = mA & 31;
                #pragma unroll
                for (int v = 0; v < 8; ++v) {
                    int K0 = 32*(v>>2) + (hi<<4) + ((v&3)<<2);
                    int tap = K0 >> 3;
                    A[u][v] = (int)a1w[(((cy + tap/3)*34 + (cx + tap%3))*8 + (K0&7)) >> 2];
                }
            }
            #pragma unroll
            for (int u = 0; u < 4; ++u) {
                v8i Z = {0,0,0,0,0,0,0,0};
                acc[u] = __builtin_amdgcn_wmma_i32_16x16x64_iu8(false, A[u], true, bf[0], Z, false, false);
            }
            #pragma unroll
            for (int u = 0; u < 4; ++u) {
                int mA = (t0+u)*16 + mlo;
                int cy = mA >> 5, cx = mA & 31;
                #pragma unroll
                for (int v = 0; v < 8; ++v) {
                    int K0 = 32*(v>>2) + (hi<<4) + ((v&3)<<2);
                    A[u][v] = (K0 < 8) ? (int)a1w[(((cy+2)*34 + (cx+2))*8 + (K0&7)) >> 2] : 0;
                }
            }
            #pragma unroll
            for (int u = 0; u < 4; ++u)
                acc[u] = __builtin_amdgcn_wmma_i32_16x16x64_iu8(false, A[u], true, bf[1], acc[u], false, false);
            #pragma unroll
            for (int u = 0; u < 4; ++u)
                #pragma unroll
                for (int r = 0; r < 8; ++r) {
                    int mm = (t0+u)*16 + r + (hi << 3);
                    float val = (float)acc[u][r]*sc + sh;
                    p2[mm*16 + ncol] = (val > 0.f) ? 1 : 0;
                }
        }
    }
    __syncthreads();

    {
        const unsigned char* p2 = smem + OFF_P2;
        unsigned char* a2 = smem + OFF_ACT2;
        for (int i = tid; i < 4096; i += 256) {
            int c = i & 15, px = (i >> 4) & 15, py = i >> 8;
            unsigned bit = 0;
            for (int a = 0; a < 2; ++a)
                for (int b = 0; b < 2; ++b)
                    bit |= p2[(((2*py+a) << 5) + 2*px+b)*16 + c];
            a2[((py+1)*18 + (px+1))*16 + c] = (unsigned char)bit;
        }
    }
    __syncthreads();

    {
        v8i bf3[2][3];
        for (int nt = 0; nt < 2; ++nt)
            for (int q = 0; q < 3; ++q)
                for (int v = 0; v < 8; ++v) {
                    unsigned d = 0;
                    for (int j = 0; j < 4; ++j) {
                        int k = 64*q + 16*(2*(v>>2) + hi) + ((v&3)<<2) + j;
                        int val = 0;
                        if (k < 144) {
                            int tap = k >> 4, cin = k & 15;
                            float w = w3[(nt*16 + ncol)*144 + cin*9 + tap];
                            val = (w>0.f)-(w<0.f);
                        }
                        d |= ((unsigned)(val & 0xFF)) << (8*j);
                    }
                    bf3[nt][q][v] = (int)d;
                }

        const unsigned* a2w = (const unsigned*)(smem + OFF_ACT2);
        float* p3 = (float*)(smem + OFF_P3);
        v8i acc[4];
        #pragma unroll
        for (int u = 0; u < 4; ++u) acc[u] = (v8i){0,0,0,0,0,0,0,0};
        for (int q = 0; q < 3; ++q) {
            v8i Aq[2];
            #pragma unroll
            for (int ml = 0; ml < 2; ++ml) {
                int mA = (wave*2 + ml)*16 + mlo;
                int cy = mA >> 4, cx = mA & 15;
                #pragma unroll
                for (int v = 0; v < 8; ++v) {
                    int k = 64*q + 32*(v>>2) + (hi<<4) + ((v&3)<<2);
                    int val = 0;
                    if (k < 144) {
                        int tap = k >> 4, c0 = k & 15;
                        val = (int)a2w[(((cy + tap/3)*18 + (cx + tap%3))*16 + c0) >> 2];
                    }
                    Aq[ml][v] = val;
                }
            }
            #pragma unroll
            for (int ml = 0; ml < 2; ++ml)
                #pragma unroll
                for (int nt = 0; nt < 2; ++nt)
                    acc[ml*2+nt] = __builtin_amdgcn_wmma_i32_16x16x64_iu8(
                        false, Aq[ml], true, bf3[nt][q], acc[ml*2+nt], false, false);
        }
        #pragma unroll
        for (int ml = 0; ml < 2; ++ml)
            #pragma unroll
            for (int nt = 0; nt < 2; ++nt) {
                int n3 = nt*16 + ncol;
                float sc3 = scale3[n3], sh3 = shift3[n3];
                #pragma unroll
                for (int r = 0; r < 8; ++r) {
                    int mm = (wave*2 + ml)*16 + r + (hi << 3);
                    float val = (float)acc[ml*2+nt][r]*sc3 + sh3;
                    p3[mm*32 + n3] = fmaxf(val, 0.f);
                }
            }
    }
    __syncthreads();

    {
        const float* p3 = (const float*)(smem + OFF_P3);
        float* h = (float*)(smem + OFF_H);
        for (int i = tid; i < 2048; i += 256) {
            int px = i & 7, py = (i >> 3) & 7, c = i >> 6;
            float mx = p3[((2*py)*16 + 2*px)*32 + c];
            mx = fmaxf(mx, p3[((2*py)*16 + 2*px+1)*32 + c]);
            mx = fmaxf(mx, p3[((2*py+1)*16 + 2*px)*32 + c]);
            mx = fmaxf(mx, p3[((2*py+1)*16 + 2*px+1)*32 + c]);
            h[i] = mx;
        }
    }
    __syncthreads();

    {
        const float* h = (const float*)(smem + OFF_H);
        float* red = (float*)(smem + OFF_RED);
        float s0 = 0.f, s1 = 0.f;
        for (int j = tid; j < 2048; j += 256) {
            float hv = h[j];
            s0 += hv * fcw[j];
            s1 += hv * fcw[2048 + j];
        }
        red[tid] = s0; red[256 + tid] = s1;
        __syncthreads();
        for (int s = 128; s > 0; s >>= 1) {
            if (tid < s) { red[tid] += red[tid + s]; red[256 + tid] += red[256 + tid + s]; }
            __syncthreads();
        }
        if (tid == 0) {
            float l0 = red[0] + fcb[0], l1 = red[256] + fcb[1];
            float mx = fmaxf(l0, l1);
            float e0 = expf(l0 - mx), e1 = expf(l1 - mx);
            float inv = 1.f / (e0 + e1);
            out[(size_t)img*2 + 0] = e0 * inv;
            out[(size_t)img*2 + 1] = e1 * inv;
        }
    }
}

extern "C" void kernel_launch(void* const* d_in, const int* in_sizes, int n_in,
                              void* d_out, int out_size, void* d_ws, size_t ws_size,
                              hipStream_t stream) {
    const float* x   = (const float*)d_in[0];
    const float* w1  = (const float*)d_in[1];
    const float* b1  = (const float*)d_in[2];
    const float* g1  = (const float*)d_in[3];
    const float* be1 = (const float*)d_in[4];
    const float* m1  = (const float*)d_in[5];
    const float* v1  = (const float*)d_in[6];
    const float* w2  = (const float*)d_in[7];
    const float* b2  = (const float*)d_in[8];
    const float* g2  = (const float*)d_in[9];
    const float* be2 = (const float*)d_in[10];
    const float* m2  = (const float*)d_in[11];
    const float* v2  = (const float*)d_in[12];
    const float* w3  = (const float*)d_in[13];
    const float* b3  = (const float*)d_in[14];
    const float* g3  = (const float*)d_in[15];
    const float* be3 = (const float*)d_in[16];
    const float* m3  = (const float*)d_in[17];
    const float* v3  = (const float*)d_in[18];
    const float* fcw = (const float*)d_in[19];
    const float* fcb = (const float*)d_in[20];
    float* out = (float*)d_out;

    const int B = in_sizes[0] / 4096;
    drone1bit_fused<<<dim3(B), dim3(256), 0, stream>>>(
        x, w1, b1, g1, be1, m1, v1,
        w2, b2, g2, be2, m2, v2,
        w3, b3, g3, be3, m3, v3,
        fcw, fcb, out);
}
